// SlidingTileAttention3D_30477087932849
// MI455X (gfx1250) — compile-verified, matrix operand order checked by reading, not run on hardware
//
#include <hip/hip_runtime.h>

typedef unsigned short u16;
typedef __attribute__((ext_vector_type(16))) __bf16 v16bf;
typedef __attribute__((ext_vector_type(8)))  __bf16 v8bf;
typedef __attribute__((ext_vector_type(8)))  float  v8f;

// ---------------- problem constants ----------------
#define HIDDEN   1024
#define SEQ      4608      // 8*24*24
#define HEADS    16
#define DH       64
#define N_TILES  36        // 4*3*3
#define TILE     128       // 2*8*8
#define NNBR     27        // 3*3*3 neighbor tiles
// tile grid: NT=4, NH=3, NW=3 ; tile dims TT=2, TH=8, TW=8

// ---------------- helpers ----------------
__device__ __forceinline__ u16 f2bf(float f) {
    unsigned u = __float_as_uint(f);
    unsigned r = u + 0x7FFFu + ((u >> 16) & 1u);   // round-to-nearest-even
    return (u16)(r >> 16);
}

__device__ __forceinline__ v8f v8f_zero() {
    v8f r;
#pragma unroll
    for (int i = 0; i < 8; ++i) r[i] = 0.0f;
    return r;
}

// Load one 16x32 (MxK) bf16 WMMA fragment for this lane from a row of bf16
// data (row-major, contiguous K). ISA layout: lane holds row m = lane%16,
// elements K = [k0 + lg*8 .. +7] then [k0 + 16 + lg*8 .. +7], lg = lane/16.
__device__ __forceinline__ v16bf frag_ld(const u16* row, int k0, int lg) {
    v8bf lo = *reinterpret_cast<const v8bf*>(row + k0 + lg * 8);
    v8bf hi = *reinterpret_cast<const v8bf*>(row + k0 + 16 + lg * 8);
    return __builtin_shufflevector(lo, hi, 0,1,2,3,4,5,6,7,8,9,10,11,12,13,14,15);
}

__device__ __forceinline__ v8f wmma_bf16(v16bf a, v16bf b, v8f c) {
    return __builtin_amdgcn_wmma_f32_16x16x32_bf16(false, a, false, b,
                                                   (short)0, c, false, false);
}

// ---------------- fp32 -> bf16 conversion ----------------
__global__ void cvt_bf16_kernel(const float* __restrict__ in,
                                u16* __restrict__ out, int n) {
    int i = blockIdx.x * blockDim.x + threadIdx.x;
    if (i < n) out[i] = f2bf(in[i]);
}

// ---------------- GEMM: out[M,N] = A[M,K] * W[N,K]^T, K=N=1024 -------------
// block = 256 threads (8 waves). wave w -> rows [blockIdx.y*256 + w*32, +32),
// cols [blockIdx.x*64, +64). Each B fragment feeds 2 WMMAs (M reuse).
__global__ __launch_bounds__(256)
void gemm_bf16_kernel(const u16* __restrict__ A, const u16* __restrict__ W,
                      u16* __restrict__ outb, float* __restrict__ outf) {
    const int lane = threadIdx.x & 31;
    const int wave = threadIdx.x >> 5;
    const int lg = lane >> 4;
    const int ln = lane & 15;
    const int n0 = blockIdx.x * 64;
    const int m0 = blockIdx.y * 256 + wave * 32;

    const u16* arow0 = A + (size_t)(m0 + ln) * HIDDEN;
    const u16* arow1 = A + (size_t)(m0 + 16 + ln) * HIDDEN;

    v8f acc[2][4];
#pragma unroll
    for (int i = 0; i < 2; ++i)
#pragma unroll
        for (int j = 0; j < 4; ++j) acc[i][j] = v8f_zero();

    for (int k0 = 0; k0 < HIDDEN; k0 += 32) {
        v16bf af0 = frag_ld(arow0, k0, lg);
        v16bf af1 = frag_ld(arow1, k0, lg);
#pragma unroll
        for (int j = 0; j < 4; ++j) {
            const u16* wrow = W + (size_t)(n0 + j * 16 + ln) * HIDDEN;
            v16bf bf_ = frag_ld(wrow, k0, lg);
            acc[0][j] = wmma_bf16(af0, bf_, acc[0][j]);
            acc[1][j] = wmma_bf16(af1, bf_, acc[1][j]);
        }
    }

    // C/D layout: lane holds col n = lane%16, row m = v + 8*lg (v = VGPR idx)
#pragma unroll
    for (int i = 0; i < 2; ++i) {
        const int mb = m0 + i * 16;
        if (outf) {
#pragma unroll
            for (int j = 0; j < 4; ++j)
#pragma unroll
                for (int v = 0; v < 8; ++v)
                    outf[(size_t)(mb + v + 8 * lg) * HIDDEN + n0 + j * 16 + ln] = acc[i][j][v];
        } else {
#pragma unroll
            for (int j = 0; j < 4; ++j)
#pragma unroll
                for (int v = 0; v < 8; ++v)
                    outb[(size_t)(mb + v + 8 * lg) * HIDDEN + n0 + j * 16 + ln] = f2bf(acc[i][j][v]);
        }
    }
}

// ---------------- sliding-tile flash attention ----------------
// grid = (36 tiles, 16 heads), block = 256 (8 waves). Wave w owns Q rows
// [w*16, w*16+16) of the 128-row tile. Streams 27 neighbor KV tiles.
__global__ __launch_bounds__(256)
void attn_kernel(const u16* __restrict__ Q, const u16* __restrict__ K,
                 const u16* __restrict__ V, u16* __restrict__ O) {
    __shared__ __align__(16) u16 shK[128][72];      // K chunk row-major (16B-aligned rows)
    __shared__ __align__(16) u16 shV[64][136];      // V^T chunk (d x kv)
    __shared__ __align__(16) u16 shP[8][16][136];   // per-wave P (16 x 128) bf16

    const int n    = blockIdx.x;            // tile index 0..35
    const int head = blockIdx.y;
    const int nt = n / 9, nh = (n % 9) / 3, nw = n % 3;
    const int ct = min(max(nt, 1), 2);      // clipped t-center (NT=4, half=1)

    const int tid  = threadIdx.x;
    const int lane = tid & 31;
    const int wave = tid >> 5;
    const int lg = lane >> 4;
    const int ln = lane & 15;
    const int colbase = head * DH;

    // ---- load Q fragments (rows wave*16 + ln), kept in registers ----
    const int qr = wave * 16 + ln;
    int tt = qr >> 6, th = (qr >> 3) & 7, tw = qr & 7;
    int qs = (nt * 2 + tt) * 576 + (nh * 8 + th) * 24 + (nw * 8 + tw);
    const u16* qptr = Q + (size_t)qs * HIDDEN + colbase;
    v16bf qf0 = frag_ld(qptr, 0, lg);
    v16bf qf1 = frag_ld(qptr, 32, lg);

    // staging decomposition (constant across chunks)
    const int kr  = tid >> 1;               // K: kv row 0..127
    const int kch = (tid & 1) * 32;         // K: d half
    const unsigned ldsK = (unsigned)(unsigned long long)&shK[kr][kch];
    const int vR  = (tid >> 3) * 4;         // V: kv rows vR..vR+3
    const int vdc = (tid & 7) * 8;          // V: d columns vdc..vdc+7

    float rm[8], rs[8];
    v8f oacc[4];
#pragma unroll
    for (int v = 0; v < 8; ++v) { rm[v] = -1e30f; rs[v] = 0.0f; }
#pragma unroll
    for (int j = 0; j < 4; ++j) oacc[j] = v8f_zero();

    const float scale = 0.125f;  // 1/sqrt(64)

    for (int j = 0; j < NNBR; ++j) {
        const int jt = j / 9, jh = (j % 9) / 3, jw = j % 3;
        const int t_t = ct + jt - 1;        // h/w centers are always 1 -> jh, jw

        // ---- K: async global->LDS copy (ASYNCcnt), one kv row half per thread
        {
            int rtt = kr >> 6, rth = (kr >> 3) & 7, rtw = kr & 7;
            int s = (t_t * 2 + rtt) * 576 + (jh * 8 + rth) * 24 + (jw * 8 + rtw);
            unsigned long long gk =
                (unsigned long long)(K + (size_t)s * HIDDEN + colbase + kch);
            asm volatile(
                "global_load_async_to_lds_b128 %0, %1, off\n\t"
                "global_load_async_to_lds_b128 %0, %1, off offset:16\n\t"
                "global_load_async_to_lds_b128 %0, %1, off offset:32\n\t"
                "global_load_async_to_lds_b128 %0, %1, off offset:48"
                :: "v"(ldsK), "v"(gk) : "memory");
        }

        // ---- V: gather 4 kv-rows x 8 d-cols, store transposed as packed b64
        {
            union { uint4 q; u16 h[8]; } vb[4];
#pragma unroll
            for (int rr = 0; rr < 4; ++rr) {
                int r = vR + rr;
                int rtt = r >> 6, rth = (r >> 3) & 7, rtw = r & 7;
                int s = (t_t * 2 + rtt) * 576 + (jh * 8 + rth) * 24 + (jw * 8 + rtw);
                vb[rr].q = *reinterpret_cast<const uint4*>(
                    V + (size_t)s * HIDDEN + colbase + vdc);
            }
#pragma unroll
            for (int d = 0; d < 8; ++d) {
                unsigned long long pk =
                    (unsigned long long)vb[0].h[d]
                  | ((unsigned long long)vb[1].h[d] << 16)
                  | ((unsigned long long)vb[2].h[d] << 32)
                  | ((unsigned long long)vb[3].h[d] << 48);
                *reinterpret_cast<unsigned long long*>(&shV[vdc + d][vR]) = pk;
            }
        }

        asm volatile("s_wait_asynccnt 0x0" ::: "memory");
        __syncthreads();

        // ---- S = (Q*scale) @ K^T : 8 N-frags x 2 K-steps ----
        v8f sacc[8];
#pragma unroll
        for (int nf = 0; nf < 8; ++nf) sacc[nf] = v8f_zero();
#pragma unroll
        for (int nf = 0; nf < 8; ++nf) {
            const u16* krow2 = &shK[nf * 16 + ln][0];
            v16bf kf0 = frag_ld(krow2, 0, lg);
            v16bf kf1 = frag_ld(krow2, 32, lg);
            sacc[nf] = wmma_bf16(qf0, kf0, sacc[nf]);
            sacc[nf] = wmma_bf16(qf1, kf1, sacc[nf]);
        }

        // ---- online softmax (row m = v + 8*lg lives in 16 lanes of half lg) --
#pragma unroll
        for (int v = 0; v < 8; ++v) {
            float cm = -1e30f;
#pragma unroll
            for (int nf = 0; nf < 8; ++nf) {
                float x = sacc[nf][v] * scale;
                sacc[nf][v] = x;
                cm = fmaxf(cm, x);
            }
#pragma unroll
            for (int off = 1; off < 16; off <<= 1)
                cm = fmaxf(cm, __shfl_xor(cm, off, 32));
            float mnew = fmaxf(rm[v], cm);
            float corr = __expf(rm[v] - mnew);
            rm[v] = mnew;
            rs[v] *= corr;
#pragma unroll
            for (int q = 0; q < 4; ++q) oacc[q][v] *= corr;
            float lsum = 0.0f;
#pragma unroll
            for (int nf = 0; nf < 8; ++nf) {
                float p = __expf(sacc[nf][v] - mnew);
                sacc[nf][v] = p;
                lsum += p;
            }
#pragma unroll
            for (int off = 1; off < 16; off <<= 1)
                lsum += __shfl_xor(lsum, off, 32);
            rs[v] += lsum;
        }

        // ---- dump P (bf16) to per-wave LDS, re-read in A-layout ----
#pragma unroll
        for (int nf = 0; nf < 8; ++nf)
#pragma unroll
            for (int v = 0; v < 8; ++v)
                shP[wave][v + 8 * lg][nf * 16 + ln] = f2bf(sacc[nf][v]);
        __syncthreads();

        // ---- O += P @ V : 4 K-steps (kv) x 4 N-frags (d) ----
#pragma unroll
        for (int ks = 0; ks < 4; ++ks) {
            const u16* prow = &shP[wave][ln][0];
            v16bf pf = frag_ld(prow, ks * 32, lg);
#pragma unroll
            for (int d4 = 0; d4 < 4; ++d4) {
                const u16* vr = &shV[d4 * 16 + ln][0];
                v16bf vf = frag_ld(vr, ks * 32, lg);
                oacc[d4] = wmma_bf16(pf, vf, oacc[d4]);
            }
        }
        __syncthreads();
    }

    // ---- normalize + scatter to [SEQ, HIDDEN] (untileize) ----
#pragma unroll
    for (int v = 0; v < 8; ++v) {
        int q = wave * 16 + v + 8 * lg;
        int qtt = q >> 6, qth = (q >> 3) & 7, qtw = q & 7;
        int s = (nt * 2 + qtt) * 576 + (nh * 8 + qth) * 24 + (nw * 8 + qtw);
        float inv = 1.0f / rs[v];
#pragma unroll
        for (int d4 = 0; d4 < 4; ++d4)
            O[(size_t)s * HIDDEN + colbase + d4 * 16 + ln] = f2bf(oacc[d4][v] * inv);
    }
}

// ---------------- host launcher ----------------
extern "C" void kernel_launch(void* const* d_in, const int* in_sizes, int n_in,
                              void* d_out, int out_size, void* d_ws, size_t ws_size,
                              hipStream_t stream) {
    const float* hs = (const float*)d_in[0];
    const float* qw = (const float*)d_in[1];
    const float* kw = (const float*)d_in[2];
    const float* vw = (const float*)d_in[3];
    const float* ow = (const float*)d_in[4];
    float* out = (float*)d_out;

    char* ws = (char*)d_ws;
    const size_t actBytes = (size_t)SEQ * HIDDEN * sizeof(u16);      // 9.44 MB
    const size_t wBytes   = (size_t)HIDDEN * HIDDEN * sizeof(u16);   // 2.10 MB
    u16* hb  = (u16*)(ws);                       // hidden bf16
    u16* qwb = (u16*)(ws + actBytes);
    u16* kwb = (u16*)(ws + actBytes + wBytes);
    u16* vwb = (u16*)(ws + actBytes + 2 * wBytes);
    u16* owb = (u16*)(ws + actBytes + 3 * wBytes);
    u16* Qb  = (u16*)(ws + actBytes + 4 * wBytes);
    u16* Kb  = (u16*)(ws + 2 * actBytes + 4 * wBytes);
    u16* Vb  = (u16*)(ws + 3 * actBytes + 4 * wBytes);
    u16* Ab  = (u16*)(ws + 4 * actBytes + 4 * wBytes);               // attn out

    // fp32 -> bf16
    {
        int n = SEQ * HIDDEN;
        cvt_bf16_kernel<<<(n + 255) / 256, 256, 0, stream>>>(hs, hb, n);
        n = HIDDEN * HIDDEN;
        cvt_bf16_kernel<<<(n + 255) / 256, 256, 0, stream>>>(qw, qwb, n);
        cvt_bf16_kernel<<<(n + 255) / 256, 256, 0, stream>>>(kw, kwb, n);
        cvt_bf16_kernel<<<(n + 255) / 256, 256, 0, stream>>>(vw, vwb, n);
        cvt_bf16_kernel<<<(n + 255) / 256, 256, 0, stream>>>(ow, owb, n);
    }

    dim3 gg(HIDDEN / 64, SEQ / 256);   // (16, 18)
    gemm_bf16_kernel<<<gg, 256, 0, stream>>>(hb, qwb, Qb, nullptr);
    gemm_bf16_kernel<<<gg, 256, 0, stream>>>(hb, kwb, Kb, nullptr);
    gemm_bf16_kernel<<<gg, 256, 0, stream>>>(hb, vwb, Vb, nullptr);

    attn_kernel<<<dim3(N_TILES, HEADS), 256, 0, stream>>>(Qb, Kb, Vb, Ab);

    gemm_bf16_kernel<<<gg, 256, 0, stream>>>(Ab, owb, nullptr, out);
}
